// WindowAttention_4922032521704
// MI455X (gfx1250) — compile-verified, matrix operand order checked by reading, not run on hardware
//
#include <hip/hip_runtime.h>
#include <hip/hip_bf16.h>

// -------------------------------------------------------------------------
// Swin window attention on gfx1250 via v_wmma_f32_16x16x32_f16 with
// double-buffered GLOBAL_LOAD_ASYNC_TO_LDS_B128 staging (ASYNCcnt) in the
// GEMMs. Shapes: B_=1024 windows, N=64 tokens, C=512, H=16 heads, hd=32.
// Pipeline: f32->f16 convert, WMMA NT-GEMM for QKV, fused per-(window,head)
// attention (QK^T -> bias+softmax -> PV, all WMMA), WMMA NT-GEMM for proj.
// -------------------------------------------------------------------------

typedef _Float16 v16h __attribute__((ext_vector_type(16)));
typedef _Float16 v8h  __attribute__((ext_vector_type(8)));
typedef float    v8f  __attribute__((ext_vector_type(8)));

union HFrag {
    v16h v;
    v8h  h[2];
};

#define B_WIN  1024
#define NTOK   64
#define CDIM   512
#define NHEAD  16
#define HDIM   32
#define QKVN   1536   // 3*C
#define MROWS  65536  // B_ * N

// ---- f32 -> f16 conversion ----------------------------------------------
__global__ void cvt_f16_kernel(const float* __restrict__ src,
                               _Float16* __restrict__ dst, long n) {
    long i = (long)blockIdx.x * blockDim.x + threadIdx.x;
    if (i < n) dst[i] = (_Float16)src[i];
}

// ---- CDNA5 async global->LDS copy (tracked by ASYNCcnt) ------------------
// addrspace(3)->generic cast places the LDS byte offset in the low 32 bits,
// which is exactly what the instruction's LDS-address VGPR expects.
__device__ __forceinline__ unsigned lds_off_of(const void* p) {
    return (unsigned)(uintptr_t)p;
}

__device__ __forceinline__ void async_copy_b128(unsigned lds_off,
                                                const void* gaddr) {
    asm volatile("global_load_async_to_lds_b128 %0, %1, off"
                 :: "v"(lds_off), "v"(gaddr)
                 : "memory");
}

__device__ __forceinline__ void wait_asynccnt_3() {
    asm volatile("s_wait_asynccnt 3" ::: "memory");
}
__device__ __forceinline__ void wait_asynccnt_0() {
    asm volatile("s_wait_asynccnt 0" ::: "memory");
}

// ---- fragment loader: row-major [rows, ld] f16, tile row `row0`, K-chunk
// `kb` (multiple of 32). Lane pattern per CDNA5 ISA 16-bit A/B layout:
// lanes 0-15 : M/N = row0+lane,    K = kb+{0..7, 16..23}
// lanes 16-31: M/N = row0+lane-16, K = kb+{8..15, 24..31}
__device__ __forceinline__ v16h load_frag_nt(const _Float16* base,
                                             int ld, int row0, int kb) {
    const int lane = threadIdx.x & 31;
    const int r    = row0 + (lane & 15);
    const int k    = kb + ((lane >> 4) << 3);
    const _Float16* p = base + (size_t)r * ld + k;
    HFrag f;
    f.h[0] = *(const v8h*)(p);
    f.h[1] = *(const v8h*)(p + 16);
    return f.v;
}

__device__ __forceinline__ v8f wmma_f16(v16h a, v16h b, v8f c) {
    return __builtin_amdgcn_wmma_f32_16x16x32_f16(
        /*neg_a=*/false, a, /*neg_b=*/false, b,
        /*c_mod=*/(short)0, c, /*reuse_a=*/false, /*reuse_b=*/false);
}

// ---- generic NT GEMM: C[m,n] = sum_k A[m,k]*B[n,k] + bias[n] -------------
// A: [M,K] f16 row-major, B: [N,K] f16 row-major, C: [M,N] OutT row-major.
// Block = 256 threads = 8 waves; block tile 64(M) x 128(N); K-step 32.
// A/B tiles staged to LDS with double-buffered async b128 copies; waves
// then compute 32x32 sub-tiles from LDS (2x2 WMMA accumulators).
template <typename OutT>
__global__ void gemm_nt_kernel(const _Float16* __restrict__ A,
                               const _Float16* __restrict__ B,
                               const float* __restrict__ bias,
                               OutT* __restrict__ C,
                               int M, int N, int K) {
    __shared__ _Float16 ldsA[2][64 * 32];    // 4 KB per buffer
    __shared__ _Float16 ldsB[2][128 * 32];   // 8 KB per buffer

    const int tid  = threadIdx.x;
    const int lane = tid & 31;
    const int wid  = tid >> 5;        // 0..7
    const int l15  = lane & 15;
    const int hi   = lane >> 4;
    const int wm   = wid & 1;         // 2 waves along M
    const int wn   = wid >> 1;        // 4 waves along N
    const int mBlk = blockIdx.x * 64;
    const int nBlk = blockIdx.y * 128;

    // staging coords: one b128/thread for A (64 rows x 64B),
    // two b128/thread for B (128 rows x 64B)
    const int arow = tid >> 2;
    const int achk = (tid & 3) * 8;   // halves

    auto stage = [&](int kb, int buf) {
        {
            const _Float16* g = A + (size_t)(mBlk + arow) * K + kb + achk;
            async_copy_b128(lds_off_of(&ldsA[buf][arow * 32 + achk]), g);
        }
#pragma unroll
        for (int j = 0; j < 2; ++j) {
            const int e   = tid + j * 256;
            const int row = e >> 2;
            const int chk = (e & 3) * 8;
            const _Float16* g = B + (size_t)(nBlk + row) * K + kb + chk;
            async_copy_b128(lds_off_of(&ldsB[buf][row * 32 + chk]), g);
        }
    };

    v8f acc[2][2];
    acc[0][0] = {}; acc[0][1] = {}; acc[1][0] = {}; acc[1][1] = {};

    const int nk = K >> 5;            // K / 32
    stage(0, 0);
    for (int t = 0; t < nk; ++t) {
        if (t + 1 < nk) {
            stage((t + 1) << 5, (t + 1) & 1);
            wait_asynccnt_3();        // next tile (3 async ops) still in flight
        } else {
            wait_asynccnt_0();
        }
        __syncthreads();

        const _Float16* a = ldsA[t & 1];
        const _Float16* b = ldsB[t & 1];
        v16h af[2], bf[2];
#pragma unroll
        for (int mt = 0; mt < 2; ++mt)
            af[mt] = load_frag_nt(a, 32, wm * 32 + mt * 16, 0);
#pragma unroll
        for (int nt = 0; nt < 2; ++nt)
            bf[nt] = load_frag_nt(b, 32, wn * 32 + nt * 16, 0);
#pragma unroll
        for (int mt = 0; mt < 2; ++mt)
#pragma unroll
            for (int nt = 0; nt < 2; ++nt)
                acc[mt][nt] = wmma_f16(af[mt], bf[nt], acc[mt][nt]);

        __syncthreads();              // done reading buf before next overwrite
    }

    float bv[2];
#pragma unroll
    for (int nt = 0; nt < 2; ++nt)
        bv[nt] = bias[nBlk + wn * 32 + nt * 16 + l15];

    // D layout: lane holds column n = n0 + l15; VGPR r holds row r + 8*hi.
#pragma unroll
    for (int mt = 0; mt < 2; ++mt)
#pragma unroll
        for (int nt = 0; nt < 2; ++nt)
#pragma unroll
            for (int r = 0; r < 8; ++r) {
                const int row = mBlk + wm * 32 + mt * 16 + r + 8 * hi;
                const int col = nBlk + wn * 32 + nt * 16 + l15;
                C[(size_t)row * N + col] = (OutT)(acc[mt][nt][r] + bv[nt]);
            }
}

// ---- fused attention per (window, head) ----------------------------------
// Block = 64 threads (2 waves); wave w owns query rows [w*32, w*32+32).
// QK^T: K=32 -> one WMMA per 16x16 tile. Softmax rows reduced with
// __shfl_xor within 16-lane halves (lanes 0-15 and 16-31 hold rows m, m+8).
__global__ void attn_kernel(const _Float16* __restrict__ qkv,
                            const float* __restrict__ bias_table,
                            _Float16* __restrict__ ctx) {
    const int bh   = blockIdx.x;     // window*16 + head
    const int b    = bh >> 4;
    const int h    = bh & 15;
    const int tid  = threadIdx.x;
    const int lane = tid & 31;
    const int wave = tid >> 5;       // 0..1
    const int l15  = lane & 15;
    const int hi   = lane >> 4;

    __shared__ _Float16 lds_p[64 * 72];   // softmaxed P, padded rows (144B)
    __shared__ _Float16 lds_vt[32 * 72];  // V transposed [hd][tokens]

    const _Float16* qbase = qkv + (size_t)b * NTOK * QKVN + h * HDIM;
    const _Float16* kbase = qbase + CDIM;
    const _Float16* vbase = qbase + 2 * CDIM;

    // stage V^T into LDS: lds_vt[d][t] = V[t][d]
    for (int e = tid; e < NTOK * HDIM; e += 64) {
        const int t = e >> 5;
        const int d = e & 31;
        lds_vt[d * 72 + t] = vbase[(size_t)t * QKVN + d];
    }

    const int m0 = wave * 32;

    // ---- S = Q K^T (wave rows m0..m0+31, all 64 key columns) ----
    v16h qf[2], kf[4];
#pragma unroll
    for (int mt = 0; mt < 2; ++mt)
        qf[mt] = load_frag_nt(qbase, QKVN, m0 + mt * 16, 0);
#pragma unroll
    for (int nt = 0; nt < 4; ++nt)
        kf[nt] = load_frag_nt(kbase, QKVN, nt * 16, 0);

    v8f s[2][4];
#pragma unroll
    for (int mt = 0; mt < 2; ++mt)
#pragma unroll
        for (int nt = 0; nt < 4; ++nt) {
            v8f z = {};
            s[mt][nt] = wmma_f16(qf[mt], kf[nt], z);
        }

    // ---- scale + relative-position bias + row softmax ----
    const float scale = 0.17677669529663687f;  // 32^-0.5
#pragma unroll
    for (int mt = 0; mt < 2; ++mt) {
#pragma unroll
        for (int r = 0; r < 8; ++r) {
            const int i  = m0 + mt * 16 + r + 8 * hi;  // query token
            const int iy = i >> 3, ix = i & 7;
            float vals[4];
            float mx = -1e30f;
#pragma unroll
            for (int nt = 0; nt < 4; ++nt) {
                const int j  = nt * 16 + l15;          // key token
                const int jy = j >> 3, jx = j & 7;
                const int rel = (iy - jy + 7) * 15 + (ix - jx + 7);
                float v = s[mt][nt][r] * scale + bias_table[rel * NHEAD + h];
                vals[nt] = v;
                mx = fmaxf(mx, v);
            }
#pragma unroll
            for (int msk = 1; msk < 16; msk <<= 1)
                mx = fmaxf(mx, __shfl_xor(mx, msk, 32));
            float sum = 0.f;
#pragma unroll
            for (int nt = 0; nt < 4; ++nt) {
                vals[nt] = __expf(vals[nt] - mx);
                sum += vals[nt];
            }
#pragma unroll
            for (int msk = 1; msk < 16; msk <<= 1)
                sum += __shfl_xor(sum, msk, 32);
            const float inv = 1.0f / sum;
#pragma unroll
            for (int nt = 0; nt < 4; ++nt)
                lds_p[i * 72 + nt * 16 + l15] = (_Float16)(vals[nt] * inv);
        }
    }

    __syncthreads();  // V^T staged and all P rows written

    // ---- O = P V : A = P[64,64] (own rows), B = V^T[32,64], K = 64 ----
    v8f o[2][2];
    o[0][0] = {}; o[0][1] = {}; o[1][0] = {}; o[1][1] = {};
#pragma unroll
    for (int ks = 0; ks < 2; ++ks) {
        const int k0 = ks * 32;
        v16h pf[2], vf[2];
#pragma unroll
        for (int mt = 0; mt < 2; ++mt)
            pf[mt] = load_frag_nt(lds_p, 72, m0 + mt * 16, k0);
#pragma unroll
        for (int nt = 0; nt < 2; ++nt)
            vf[nt] = load_frag_nt(lds_vt, 72, nt * 16, k0);
#pragma unroll
        for (int mt = 0; mt < 2; ++mt)
#pragma unroll
            for (int nt = 0; nt < 2; ++nt)
                o[mt][nt] = wmma_f16(pf[mt], vf[nt], o[mt][nt]);
    }

    // ---- store context, layout [B_*N, C] f16 (head-major inner dim) ----
    _Float16* cb = ctx + (size_t)b * NTOK * CDIM + h * HDIM;
#pragma unroll
    for (int mt = 0; mt < 2; ++mt)
#pragma unroll
        for (int nt = 0; nt < 2; ++nt)
#pragma unroll
            for (int r = 0; r < 8; ++r) {
                const int t = m0 + mt * 16 + r + 8 * hi;
                const int d = nt * 16 + l15;
                cb[(size_t)t * CDIM + d] = (_Float16)o[mt][nt][r];
            }
}

// -------------------------------------------------------------------------
extern "C" void kernel_launch(void* const* d_in, const int* in_sizes, int n_in,
                              void* d_out, int out_size, void* d_ws, size_t ws_size,
                              hipStream_t stream) {
    const float* x          = (const float*)d_in[0];
    const float* qkv_w      = (const float*)d_in[1];
    const float* qkv_b      = (const float*)d_in[2];
    const float* proj_w     = (const float*)d_in[3];
    const float* proj_b     = (const float*)d_in[4];
    const float* bias_table = (const float*)d_in[5];
    float* out = (float*)d_out;

    // workspace carve-up (all f16), offsets keep 16B alignment
    _Float16* qkvw_h = (_Float16*)d_ws;
    _Float16* projw_h = qkvw_h + (size_t)QKVN * CDIM;            // 1536*512
    _Float16* x_h     = projw_h + (size_t)CDIM * CDIM;           // 512*512
    _Float16* qkv_h   = x_h + (size_t)MROWS * CDIM;              // 65536*512
    _Float16* ctx_h   = qkv_h + (size_t)MROWS * QKVN;            // 65536*1536

    const long n_qkvw = (long)QKVN * CDIM;
    const long n_projw = (long)CDIM * CDIM;
    const long n_x = (long)MROWS * CDIM;

    cvt_f16_kernel<<<dim3((n_qkvw + 255) / 256), dim3(256), 0, stream>>>(
        qkv_w, qkvw_h, n_qkvw);
    cvt_f16_kernel<<<dim3((n_projw + 255) / 256), dim3(256), 0, stream>>>(
        proj_w, projw_h, n_projw);
    cvt_f16_kernel<<<dim3((n_x + 255) / 256), dim3(256), 0, stream>>>(
        x, x_h, n_x);

    // QKV = x @ qkv_w^T + qkv_b  -> f16 [65536, 1536]
    gemm_nt_kernel<_Float16><<<dim3(MROWS / 64, QKVN / 128), dim3(256), 0, stream>>>(
        x_h, qkvw_h, qkv_b, qkv_h, MROWS, QKVN, CDIM);

    // fused attention per (window, head)
    attn_kernel<<<dim3(B_WIN * NHEAD), dim3(64), 0, stream>>>(
        qkv_h, bias_table, ctx_h);

    // out = ctx @ proj_w^T + proj_b  -> f32 [65536, 512]
    gemm_nt_kernel<float><<<dim3(MROWS / 64, CDIM / 128), dim3(256), 0, stream>>>(
        ctx_h, projw_h, proj_b, out, MROWS, CDIM, CDIM);
}
